// GPTModel_10797547782457
// MI455X (gfx1250) — compile-verified, matrix operand order checked by reading, not run on hardware
//
#include <hip/hip_runtime.h>
#include <hip/hip_bf16.h>

// ---------------- model constants ----------------
#define B_  2
#define T_  2048
#define D_  1024
#define H_  16
#define L_  6
#define V_  32000
#define HD_ 64
#define M_  (B_ * T_)      // 4096 rows of activations

// ---------------- WMMA types ----------------
typedef __attribute__((ext_vector_type(16))) __bf16 v16bf;
typedef __attribute__((ext_vector_type(8)))  float  v8f;

union FragU { v16bf v; unsigned int u[8]; uint4 q[2]; };

static __device__ __forceinline__ unsigned short f2bf(float f) {
    unsigned int u = __float_as_uint(f);
    unsigned int r = u + 0x7FFFu + ((u >> 16) & 1u);   // round-to-nearest-even
    return (unsigned short)(r >> 16);
}

// ---------------- gfx1250 async global->LDS copy (ASYNCcnt path) ----------------
#if defined(__has_builtin)
#  if __has_builtin(__builtin_amdgcn_global_load_async_to_lds_b128)
#    define HAVE_ASYNC_LDS 1
#  endif
#  if __has_builtin(__builtin_amdgcn_s_wait_asynccnt)
#    define HAVE_WAIT_ASYNC 1
#  endif
#endif

typedef int v4i_t __attribute__((vector_size(16)));
typedef __attribute__((address_space(1))) v4i_t* g_v4i_p;
typedef __attribute__((address_space(3))) v4i_t* l_v4i_p;

static __device__ __forceinline__ void async_copy16(const unsigned short* g,
                                                    unsigned short* l) {
#ifdef HAVE_ASYNC_LDS
    __builtin_amdgcn_global_load_async_to_lds_b128(
        (g_v4i_p)(void*)g, (l_v4i_p)(void*)l, 0, 0);
#else
    *reinterpret_cast<uint4*>(l) = *reinterpret_cast<const uint4*>(g);
#endif
}

static __device__ __forceinline__ void wait_async() {
#ifdef HAVE_ASYNC_LDS
#  ifdef HAVE_WAIT_ASYNC
    __builtin_amdgcn_s_wait_asynccnt(0);
#  else
    asm volatile("s_wait_asynccnt 0x0" ::: "memory");
#  endif
#endif
}

// K index inside a 16-bit A/B fragment for VGPR-pair p (0..7) and lane half.
// (ISA 7.12.2) u[0..3] = K {half*8 .. half*8+7} (contiguous 16B),
//              u[4..7] = K {16+half*8 ..} (contiguous 16B).
static __device__ __forceinline__ int frag_k(int p, int half) {
    return ((p >> 2) << 4) + (half << 3) + ((p & 3) << 1);
}

// ============================================================
// Embedding: x[b,t,:] = tok_emb[tokens[b,t]] + pos_emb[t]
// ============================================================
__global__ __launch_bounds__(256) void embed_kernel(
    const int* __restrict__ tokens, const float* __restrict__ tok_emb,
    const float* __restrict__ pos_emb, float* __restrict__ x)
{
    int row = blockIdx.x;                 // 0..M_-1
    int t   = row % T_;
    int tok = tokens[row];
    int i   = threadIdx.x * 4;
    float4 e = *reinterpret_cast<const float4*>(tok_emb + (size_t)tok * D_ + i);
    float4 p = *reinterpret_cast<const float4*>(pos_emb + (size_t)t   * D_ + i);
    float4 r; r.x = e.x + p.x; r.y = e.y + p.y; r.z = e.z + p.z; r.w = e.w + p.w;
    *reinterpret_cast<float4*>(x + (size_t)row * D_ + i) = r;
}

// ============================================================
// LayerNorm over D=1024, one row per block, bf16 output
// ============================================================
__global__ __launch_bounds__(256) void layernorm_kernel(
    const float* __restrict__ x, const float* __restrict__ g,
    const float* __restrict__ b, unsigned short* __restrict__ out)
{
    __shared__ float s1[256];
    __shared__ float s2[256];
    int row = blockIdx.x;
    int tid = threadIdx.x;
    float4 v = *reinterpret_cast<const float4*>(x + (size_t)row * D_ + tid * 4);
    float sum = v.x + v.y + v.z + v.w;
    float sq  = v.x * v.x + v.y * v.y + v.z * v.z + v.w * v.w;
    s1[tid] = sum; s2[tid] = sq;
    __syncthreads();
    for (int s = 128; s > 0; s >>= 1) {
        if (tid < s) { s1[tid] += s1[tid + s]; s2[tid] += s2[tid + s]; }
        __syncthreads();
    }
    float mean = s1[0] * (1.0f / D_);
    float var  = s2[0] * (1.0f / D_) - mean * mean;
    float inv  = rsqrtf(var + 1e-5f);
    const float* pv = &v.x;
    #pragma unroll
    for (int j = 0; j < 4; j++) {
        int c = tid * 4 + j;
        float val = (pv[j] - mean) * inv * g[c] + b[c];
        out[(size_t)row * D_ + c] = f2bf(val);
    }
}

// ============================================================
// GEMM: C[M,N] = A_bf16[M,K] @ W_f32[K,N] (+bias)(+GELU)(+residual)
// Block tile 128x128, BK=64; 8 waves each computing 32x64 via 2x4 WMMA accs.
// A tile staged via GLOBAL_LOAD_ASYNC_TO_LDS_B128; W converted f32->bf16,
// stored k-transposed so B-fragment k-pairs are contiguous b128 loads.
// ============================================================
#define BM  128
#define BN  128
#define BKK 64
#define SAS 72      // sA row stride (ushort), padded vs 64 to break bank conflicts
#define SWS 72      // sW row stride (ushort)

template <bool GELU_ACT>
__global__ __launch_bounds__(256) void gemm_bf16_kernel(
    const unsigned short* __restrict__ A, const float* __restrict__ W,
    const float* __restrict__ bias, const float* __restrict__ residual,
    float* __restrict__ outF, unsigned short* __restrict__ outB,
    int M, int N, int K)
{
    __shared__ __align__(16) unsigned short sA[BM * SAS];   // [m][k]
    __shared__ __align__(16) unsigned short sW[BN * SWS];   // [n][k] (k-transposed)

    int tid  = threadIdx.x;
    int wave = tid >> 5, lane = tid & 31;
    int mloc = lane & 15, half = lane >> 4;
    int m0 = blockIdx.y * BM, n0 = blockIdx.x * BN;
    int wm = (wave >> 1) * 32;       // 4 wave-rows of 32
    int wn = (wave & 1) * 64;        // 2 wave-cols of 64

    v8f acc[2][4];
    #pragma unroll
    for (int mi = 0; mi < 2; mi++)
        #pragma unroll
        for (int ni = 0; ni < 4; ni++) acc[mi][ni] = v8f{};

    for (int k0 = 0; k0 < K; k0 += BKK) {
        // ---- stage A tile: 128x64 bf16 via async b128 (1024 chunks / 256 thr) ----
        #pragma unroll
        for (int i = 0; i < 4; i++) {
            int c   = tid + i * 256;
            int r   = c >> 3;              // 0..127
            int col = (c & 7) << 3;        // 0..56
            async_copy16(A + (size_t)(m0 + r) * K + k0 + col,
                         &sA[r * SAS + col]);
        }
        // ---- stage W tile: 64x128 f32 -> bf16, stored [n][k] ----
        #pragma unroll
        for (int i = 0; i < 8; i++) {
            int c  = tid + i * 256;        // chunk of 4 floats, 2048 chunks
            int kk = c >> 5;               // k row 0..63
            int n4 = (c & 31) << 2;        // n col 0..124
            float4 w = *reinterpret_cast<const float4*>(
                W + (size_t)(k0 + kk) * N + n0 + n4);
            sW[(n4 + 0) * SWS + kk] = f2bf(w.x);
            sW[(n4 + 1) * SWS + kk] = f2bf(w.y);
            sW[(n4 + 2) * SWS + kk] = f2bf(w.z);
            sW[(n4 + 3) * SWS + kk] = f2bf(w.w);
        }
        if (k0 + BKK < K) {
            __builtin_prefetch(A + (size_t)(m0 + (tid >> 1)) * K + k0 + BKK, 0, 1);
            __builtin_prefetch(W + (size_t)(k0 + BKK + (tid >> 2)) * N + n0, 0, 1);
        }
        wait_async();
        __syncthreads();

        // ---- 2 k-steps of 32, 8 WMMA each ----
        #pragma unroll
        for (int ks = 0; ks < 2; ks++) {
            int kb = ks * 32;
            v16bf af[2], bf_[4];
            #pragma unroll
            for (int mi = 0; mi < 2; mi++) {
                FragU u;
                int row = wm + mi * 16 + mloc;
                u.q[0] = *reinterpret_cast<const uint4*>(&sA[row * SAS + kb + half * 8]);
                u.q[1] = *reinterpret_cast<const uint4*>(&sA[row * SAS + kb + 16 + half * 8]);
                af[mi] = u.v;
            }
            #pragma unroll
            for (int ni = 0; ni < 4; ni++) {
                FragU u;
                int nn = wn + ni * 16 + mloc;
                u.q[0] = *reinterpret_cast<const uint4*>(&sW[nn * SWS + kb + half * 8]);
                u.q[1] = *reinterpret_cast<const uint4*>(&sW[nn * SWS + kb + 16 + half * 8]);
                bf_[ni] = u.v;
            }
            #pragma unroll
            for (int mi = 0; mi < 2; mi++)
                #pragma unroll
                for (int ni = 0; ni < 4; ni++)
                    acc[mi][ni] = __builtin_amdgcn_wmma_f32_16x16x32_bf16(
                        false, af[mi], false, bf_[ni], (short)0, acc[mi][ni],
                        false, false);
        }
        __syncthreads();
    }

    // ---- epilogue: C layout row = r + 8*half, col = lane%16 ----
    #pragma unroll
    for (int mi = 0; mi < 2; mi++) {
        #pragma unroll
        for (int ni = 0; ni < 4; ni++) {
            #pragma unroll
            for (int r = 0; r < 8; r++) {
                int row = m0 + wm + mi * 16 + r + half * 8;
                int col = n0 + wn + ni * 16 + mloc;
                float v = acc[mi][ni][r];
                if (bias) v += bias[col];
                if (GELU_ACT) v = 0.5f * v * (1.0f + erff(v * 0.70710678f));
                if (residual) v += residual[(size_t)row * N + col];
                if (outF) outF[(size_t)row * N + col] = v;
                if (outB) outB[(size_t)row * N + col] = f2bf(v);
            }
        }
    }
}

// ============================================================
// Flash attention: one block = (b, h, 64 q rows); 4 waves x 16 rows.
// Q/K/V/O are bf16 [B*T, D] with head h at column h*64.
// ============================================================
#define BQ  64
#define BKV 64

__global__ __launch_bounds__(128) void flash_attn_kernel(
    const unsigned short* __restrict__ Q, const unsigned short* __restrict__ K,
    const unsigned short* __restrict__ V, unsigned short* __restrict__ O)
{
    __shared__ __align__(16) unsigned short sK[BKV * HD_];     // [key][d]
    __shared__ __align__(16) unsigned short sV[HD_ * BKV];     // [d][key]
    __shared__ __align__(16) unsigned short sP[4][16 * BKV];   // per-wave P tile

    int tid  = threadIdx.x;
    int wave = tid >> 5, lane = tid & 31;
    int mloc = lane & 15, half = lane >> 4;
    int qb = blockIdx.x * BQ;
    int b  = blockIdx.y / H_;
    int h  = blockIdx.y % H_;
    size_t base = (size_t)(b * T_) * D_ + (size_t)h * HD_;

    // load Q fragments (16 rows per wave, d split into two 32-chunks)
    v16bf qf[2];
    #pragma unroll
    for (int f = 0; f < 2; f++) {
        FragU u;
        int row = qb + wave * 16 + mloc;
        const unsigned short* qp = Q + base + (size_t)row * D_ + f * 32;
        u.q[0] = *reinterpret_cast<const uint4*>(qp + half * 8);
        u.q[1] = *reinterpret_cast<const uint4*>(qp + 16 + half * 8);
        qf[f] = u.v;
    }

    v8f o_[4];
    o_[0] = v8f{}; o_[1] = v8f{}; o_[2] = v8f{}; o_[3] = v8f{};
    float mi_[8], li_[8];
    #pragma unroll
    for (int r = 0; r < 8; r++) { mi_[r] = -1e30f; li_[r] = 0.0f; }

    const float scale = 0.125f;  // 1/sqrt(64)

    for (int kb = 0; kb <= qb; kb += BKV) {
        // ---- stage K [key][d] (async) and V transposed [d][key] ----
        #pragma unroll
        for (int i = 0; i < 4; i++) {
            int c   = tid + i * 128;        // 0..511
            int key = c >> 3;
            int col = (c & 7) << 3;
            async_copy16(K + base + (size_t)(kb + key) * D_ + col,
                         &sK[key * HD_ + col]);
            uint4 vv = *reinterpret_cast<const uint4*>(
                V + base + (size_t)(kb + key) * D_ + col);
            const unsigned short* pv = reinterpret_cast<const unsigned short*>(&vv);
            #pragma unroll
            for (int j = 0; j < 8; j++) sV[(col + j) * BKV + key] = pv[j];
        }
        wait_async();
        __syncthreads();

        // ---- S = Q @ K^T (16 x 64 per wave) ----
        v8f s[4];
        s[0] = v8f{}; s[1] = v8f{}; s[2] = v8f{}; s[3] = v8f{};
        #pragma unroll
        for (int f = 0; f < 2; f++) {
            #pragma unroll
            for (int ni = 0; ni < 4; ni++) {
                FragU u;
                const unsigned short* kp = &sK[(ni * 16 + mloc) * HD_ + f * 32];
                u.q[0] = *reinterpret_cast<const uint4*>(kp + half * 8);
                u.q[1] = *reinterpret_cast<const uint4*>(kp + 16 + half * 8);
                s[ni] = __builtin_amdgcn_wmma_f32_16x16x32_bf16(
                    false, qf[f], false, u.v, (short)0, s[ni], false, false);
            }
        }

        // ---- scale + causal mask ----
        #pragma unroll
        for (int ni = 0; ni < 4; ni++) {
            #pragma unroll
            for (int r = 0; r < 8; r++) {
                int qrow = qb + wave * 16 + r + half * 8;
                int kcol = kb + ni * 16 + mloc;
                float val = s[ni][r] * scale;
                if (kcol > qrow) val = -1e30f;
                s[ni][r] = val;
            }
        }

        // ---- online softmax (row = 16-lane half, shfl_xor reduce) ----
        float alpha[8];
        #pragma unroll
        for (int r = 0; r < 8; r++) {
            float v = fmaxf(fmaxf(s[0][r], s[1][r]), fmaxf(s[2][r], s[3][r]));
            #pragma unroll
            for (int off = 1; off < 16; off <<= 1)
                v = fmaxf(v, __shfl_xor(v, off, 32));
            float mnew = fmaxf(mi_[r], v);
            alpha[r] = __expf(mi_[r] - mnew);
            mi_[r] = mnew;
        }
        float rsum[8];
        #pragma unroll
        for (int r = 0; r < 8; r++) rsum[r] = 0.0f;
        #pragma unroll
        for (int ni = 0; ni < 4; ni++) {
            #pragma unroll
            for (int r = 0; r < 8; r++) {
                float p = __expf(s[ni][r] - mi_[r]);
                s[ni][r] = p;
                rsum[r] += p;
            }
        }
        #pragma unroll
        for (int r = 0; r < 8; r++) {
            float v = rsum[r];
            #pragma unroll
            for (int off = 1; off < 16; off <<= 1)
                v += __shfl_xor(v, off, 32);
            li_[r] = li_[r] * alpha[r] + v;
        }
        #pragma unroll
        for (int ni = 0; ni < 4; ni++)
            #pragma unroll
            for (int r = 0; r < 8; r++)
                o_[ni][r] *= alpha[r];

        // ---- P to LDS (bf16, per-wave tile) ----
        #pragma unroll
        for (int ni = 0; ni < 4; ni++)
            #pragma unroll
            for (int r = 0; r < 8; r++)
                sP[wave][(r + half * 8) * BKV + ni * 16 + mloc] = f2bf(s[ni][r]);
        __syncthreads();

        // ---- O += P @ V ----
        v16bf pf[2];
        #pragma unroll
        for (int f = 0; f < 2; f++) {
            FragU u;
            const unsigned short* pp = &sP[wave][mloc * BKV + f * 32];
            u.q[0] = *reinterpret_cast<const uint4*>(pp + half * 8);
            u.q[1] = *reinterpret_cast<const uint4*>(pp + 16 + half * 8);
            pf[f] = u.v;
        }
        #pragma unroll
        for (int ni = 0; ni < 4; ni++) {
            #pragma unroll
            for (int f = 0; f < 2; f++) {
                FragU u;
                const unsigned short* vp = &sV[(ni * 16 + mloc) * BKV + f * 32];
                u.q[0] = *reinterpret_cast<const uint4*>(vp + half * 8);
                u.q[1] = *reinterpret_cast<const uint4*>(vp + 16 + half * 8);
                o_[ni] = __builtin_amdgcn_wmma_f32_16x16x32_bf16(
                    false, pf[f], false, u.v, (short)0, o_[ni], false, false);
            }
        }
        __syncthreads();   // before next iteration overwrites sK/sV
    }

    // ---- epilogue: O /= l, write bf16 ----
    #pragma unroll
    for (int ni = 0; ni < 4; ni++) {
        #pragma unroll
        for (int r = 0; r < 8; r++) {
            int qrow = qb + wave * 16 + r + half * 8;
            float v = o_[ni][r] / li_[r];
            O[base + (size_t)qrow * D_ + ni * 16 + mloc] = f2bf(v);
        }
    }
}

// ============================================================
// Host launcher
// ============================================================
extern "C" void kernel_launch(void* const* d_in, const int* in_sizes, int n_in,
                              void* d_out, int out_size, void* d_ws, size_t ws_size,
                              hipStream_t stream) {
    const int*   tokens  = (const int*)  d_in[0];
    const float* tok_emb = (const float*)d_in[1];
    const float* pos_emb = (const float*)d_in[2];
    const float* Wq      = (const float*)d_in[3];
    const float* Wk      = (const float*)d_in[4];
    const float* Wv      = (const float*)d_in[5];
    const float* Wo      = (const float*)d_in[6];
    const float* bo      = (const float*)d_in[7];
    const float* ln1_g   = (const float*)d_in[8];
    const float* ln1_b   = (const float*)d_in[9];
    const float* ln2_g   = (const float*)d_in[10];
    const float* ln2_b   = (const float*)d_in[11];
    const float* W1      = (const float*)d_in[12];
    const float* b1      = (const float*)d_in[13];
    const float* W2      = (const float*)d_in[14];
    const float* b2      = (const float*)d_in[15];
    const float* lnf_g   = (const float*)d_in[16];
    const float* lnf_b   = (const float*)d_in[17];
    const float* Wout    = (const float*)d_in[18];
    float* out = (float*)d_out;

    char* ws = (char*)d_ws;
    const size_t MD_F32  = (size_t)M_ * D_ * sizeof(float);          // 16 MB
    const size_t MD_BF16 = (size_t)M_ * D_ * sizeof(unsigned short); //  8 MB
    float*          x    = (float*)ws;                                ws += MD_F32;
    unsigned short* hbuf = (unsigned short*)ws;                       ws += MD_BF16;
    unsigned short* qbuf = (unsigned short*)ws;                       ws += MD_BF16;
    unsigned short* kbuf = (unsigned short*)ws;                       ws += MD_BF16;
    unsigned short* vbuf = (unsigned short*)ws;                       ws += MD_BF16;
    unsigned short* abuf = (unsigned short*)ws;                       ws += MD_BF16;
    unsigned short* actb = (unsigned short*)ws;                       // 32 MB (M x 4D)

    dim3 blk256(256), blk128(128);
    dim3 gemmDD (D_ / BN,      M_ / BM);    // N = D
    dim3 gemmD4D(4 * D_ / BN,  M_ / BM);    // N = 4D
    dim3 gemmV  (V_ / BN,      M_ / BM);    // N = V (32000/128 = 250)
    dim3 flashG (T_ / BQ, B_ * H_);

    embed_kernel<<<M_, blk256, 0, stream>>>(tokens, tok_emb, pos_emb, x);

    for (int l = 0; l < L_; l++) {
        const size_t oDD = (size_t)l * D_ * D_;
        const size_t oD4 = (size_t)l * D_ * 4 * D_;

        layernorm_kernel<<<M_, blk256, 0, stream>>>(x, ln1_g + l * D_, ln1_b + l * D_, hbuf);

        gemm_bf16_kernel<false><<<gemmDD, blk256, 0, stream>>>(
            hbuf, Wq + oDD, nullptr, nullptr, nullptr, qbuf, M_, D_, D_);
        gemm_bf16_kernel<false><<<gemmDD, blk256, 0, stream>>>(
            hbuf, Wk + oDD, nullptr, nullptr, nullptr, kbuf, M_, D_, D_);
        gemm_bf16_kernel<false><<<gemmDD, blk256, 0, stream>>>(
            hbuf, Wv + oDD, nullptr, nullptr, nullptr, vbuf, M_, D_, D_);

        flash_attn_kernel<<<flashG, blk128, 0, stream>>>(qbuf, kbuf, vbuf, abuf);

        // x = x + attn_out @ Wo + bo
        gemm_bf16_kernel<false><<<gemmDD, blk256, 0, stream>>>(
            abuf, Wo + oDD, bo + l * D_, x, x, nullptr, M_, D_, D_);

        layernorm_kernel<<<M_, blk256, 0, stream>>>(x, ln2_g + l * D_, ln2_b + l * D_, hbuf);

        // act = gelu(h @ W1 + b1), bf16
        gemm_bf16_kernel<true><<<gemmD4D, blk256, 0, stream>>>(
            hbuf, W1 + oD4, b1 + (size_t)l * 4 * D_, nullptr, nullptr, actb,
            M_, 4 * D_, D_);
        // x = x + act @ W2 + b2
        gemm_bf16_kernel<false><<<gemmDD, blk256, 0, stream>>>(
            actb, W2 + oD4, b2 + l * D_, x, x, nullptr, M_, D_, 4 * D_);
    }

    layernorm_kernel<<<M_, blk256, 0, stream>>>(x, lnf_g, lnf_b, hbuf);
    gemm_bf16_kernel<false><<<gemmV, blk256, 0, stream>>>(
        hbuf, Wout, nullptr, nullptr, out, nullptr, M_, V_, D_);
}
